// Seq2SQL_62895501083186
// MI455X (gfx1250) — matrix operand order repaired and checked by reading, not compiled, not run on hardware
//
#include <hip/hip_runtime.h>
#include <hip/hip_bf16.h>

// ---------------- types ----------------
typedef __attribute__((ext_vector_type(16))) _Float16 v16h;
typedef __attribute__((ext_vector_type(8)))  _Float16 v8h;
typedef __attribute__((ext_vector_type(8)))  float    v8f;

#define DEVI static __device__ __forceinline__

// Model constants
#define BB   64
#define SS   64
#define TT   32
#define NW   300
#define NWP  320   // padded K for embedding dim
#define NH   512
#define H2c  256
#define MTOK 200

// ---------------- WMMA fragment loaders (f16 16x16x32) ----------------
// A fragment: 16x32 (MxK). lane: m = lane&15, K-half = lane>>4.
// slots 0..15 -> K = 16*(lane>>4) + 0..15
DEVI v16h frag_a(const _Float16* __restrict__ base, int ld, int lane) {
  int m  = lane & 15;
  int kh = (lane >> 4) << 4;
  const _Float16* p = base + m * ld + kh;
  v8h lo = *(const v8h*)(p);
  v8h hi = *(const v8h*)(p + 8);
  return __builtin_shufflevector(lo, hi, 0,1,2,3,4,5,6,7,8,9,10,11,12,13,14,15);
}

// B fragment for C = A * W^T : B(k,n) = W[n0+n][k0+k]
// lane: n = lane&15, k = 16*(lane>>4) + slot  -> 16 contiguous halves of row n
DEVI v16h frag_b(const _Float16* __restrict__ W, int ldw, int n0, int k0, int lane) {
  const _Float16* p = W + (size_t)(n0 + (lane & 15)) * ldw + (size_t)k0 + ((lane >> 4) << 4);
  return *(const v16h*)p;
}

DEVI float sigm(float x) { return 1.0f / (1.0f + expf(-x)); }

// ---------------- weight / activation f16 convert (with K padding) ----------------
__global__ __launch_bounds__(256)
void convert_pad_kernel(const float* __restrict__ src, _Float16* __restrict__ dst,
                        int rows, int sk, int dk) {
  int i = blockIdx.x * blockDim.x + threadIdx.x;
  if (i >= rows * dk) return;
  int r = i / dk, k = i - r * dk;
  dst[i] = (k < sk) ? (_Float16)src[(size_t)r * sk + k] : (_Float16)0.0f;
}

// ---------------- bulk GEMM: C[M,N] = A[M,K](f16) * W[N,K](f16)^T + bias ----------------
__global__ __launch_bounds__(256)
void wmma_gemm_bias_kernel(const _Float16* __restrict__ A, const _Float16* __restrict__ W,
                           const float* __restrict__ bias, float* __restrict__ C,
                           int M, int N, int K, int lda, int ldw, int ldc) {
  int lane = threadIdx.x & 31;
  int wave = threadIdx.x >> 5;
  int ntn  = N >> 4;
  int tile = blockIdx.x * 8 + wave;
  if (tile >= (M >> 4) * ntn) return;
  int mt = tile / ntn, nt = tile - mt * ntn;
  const _Float16* Abase = A + (size_t)(mt << 4) * lda;
  v8f acc = {};
  for (int k0 = 0; k0 < K; k0 += 32) {
    v16h a = frag_a(Abase + k0, lda, lane);
    v16h b = frag_b(W, ldw, nt << 4, k0, lane);
    acc = __builtin_amdgcn_wmma_f32_16x16x32_f16(false, a, false, b, (short)0, acc, false, false);
  }
  int n     = (nt << 4) + (lane & 15);
  int mbase = (mt << 4) + ((lane >> 4) << 3);
  float bv = bias ? bias[n] : 0.0f;
#pragma unroll
  for (int r = 0; r < 8; ++r)
    C[(size_t)(mbase + r) * ldc + n] = acc[r] + bv;
}

// ---------------- LSTM recurrence ----------------
// 16 batch rows per block (batch rows are independent recurrences -> no grid sync).
// blockDim = (H/16)*32 waves; wave q owns gate quadruple {q, q+H/16, q+2H/16, q+3H/16}.
// h/c state lives in VGPRs; h broadcast between steps via 16xH f16 LDS tile.
__global__ __launch_bounds__(1024)
void lstm_recur_kernel(const float* __restrict__ inG,     // [64, S, 4H] (bias folded in)
                       const _Float16* __restrict__ Whh,  // [4H, H] f16 row-major
                       const float* __restrict__ h0,      // [64,H] or nullptr
                       const float* __restrict__ c0,
                       const int* __restrict__ lens,      // [64]
                       _Float16* __restrict__ out,        // rows 64*S, stride ldo, + col0
                       int ldo, int col0,
                       float* __restrict__ hF, float* __restrict__ cF,
                       int S, int H, int reverse) {
  int lane = threadIdx.x & 31;
  int q    = threadIdx.x >> 5;
  int b0   = blockIdx.x << 4;
  int G    = H << 2;
  int n    = lane & 15;
  int mh   = (lane >> 4) << 3;
  int hh   = (q << 4) + n;
  int tI   = q << 4;

  __shared__ __attribute__((aligned(32))) _Float16 hS[16 * 512];

  float hreg[8], creg[8];
  int lenr[8];
#pragma unroll
  for (int r = 0; r < 8; ++r) {
    int m = mh + r;
    hreg[r] = h0 ? h0[(size_t)(b0 + m) * H + hh] : 0.0f;
    creg[r] = c0 ? c0[(size_t)(b0 + m) * H + hh] : 0.0f;
    lenr[r] = lens[b0 + m];
    hS[m * H + hh] = (_Float16)hreg[r];
  }
  __syncthreads();

  for (int step = 0; step < S; ++step) {
    int t = reverse ? (S - 1 - step) : step;
    v8f aI = {}, aF = {}, aG = {}, aO = {};
    for (int k0 = 0; k0 < H; k0 += 32) {
      v16h a  = frag_a(hS + k0, H, lane);
      v16h bI = frag_b(Whh, H, tI,         k0, lane);
      v16h bF = frag_b(Whh, H, tI + H,     k0, lane);
      v16h bG = frag_b(Whh, H, tI + 2 * H, k0, lane);
      v16h bO = frag_b(Whh, H, tI + 3 * H, k0, lane);
      aI = __builtin_amdgcn_wmma_f32_16x16x32_f16(false, a, false, bI, (short)0, aI, false, false);
      aF = __builtin_amdgcn_wmma_f32_16x16x32_f16(false, a, false, bF, (short)0, aF, false, false);
      aG = __builtin_amdgcn_wmma_f32_16x16x32_f16(false, a, false, bG, (short)0, aG, false, false);
      aO = __builtin_amdgcn_wmma_f32_16x16x32_f16(false, a, false, bO, (short)0, aO, false, false);
    }
    __syncthreads();  // all waves done reading hS
#pragma unroll
    for (int r = 0; r < 8; ++r) {
      int m = mh + r;
      const float* ig = inG + ((size_t)(b0 + m) * S + t) * G;
      float gi = aI[r] + ig[hh];
      float gf = aF[r] + ig[H + hh];
      float gg = aG[r] + ig[2 * H + hh];
      float go = aO[r] + ig[3 * H + hh];
      float cn = sigm(gf) * creg[r] + sigm(gi) * tanhf(gg);
      float hn = sigm(go) * tanhf(cn);
      bool valid = t < lenr[r];
      creg[r] = valid ? cn : creg[r];
      hreg[r] = valid ? hn : hreg[r];
      out[((size_t)(b0 + m) * S + t) * ldo + col0 + hh] = (_Float16)(valid ? hn : 0.0f);
      hS[m * H + hh] = (_Float16)hreg[r];
    }
    __syncthreads();  // hS ready for next step
  }
  if (hF) {
#pragma unroll
    for (int r = 0; r < 8; ++r) {
      hF[(size_t)(b0 + mh + r) * H + hh] = hreg[r];
      cF[(size_t)(b0 + mh + r) * H + hh] = creg[r];
    }
  }
}

// ---------------- decoder one-hot input: gather Wih columns + bias ----------------
__global__ __launch_bounds__(256)
void dec_gather_kernel(const float* __restrict__ Wih, const float* __restrict__ b,
                       const int* __restrict__ tok, float* __restrict__ outG) {
  int i = blockIdx.x * blockDim.x + threadIdx.x;  // over 2048(bt) * 2048(n)
  if (i >= BB * TT * 4 * NH) return;
  int bt = i >> 11;
  int n  = i & 2047;
  outG[i] = Wih[(size_t)n * MTOK + tok[bt]] + b[n];
}

// ---------------- decoder initial states: dh = concat(h, h) ----------------
__global__ __launch_bounds__(256)
void build_dec_init_kernel(const float* __restrict__ h0f, const float* __restrict__ c0f,
                           const float* __restrict__ h0b, const float* __restrict__ c0b,
                           float* __restrict__ dh0, float* __restrict__ dc0,
                           float* __restrict__ dh1, float* __restrict__ dc1) {
  int i = blockIdx.x * blockDim.x + threadIdx.x;  // 64*512
  if (i >= BB * NH) return;
  int bb = i >> 9;
  int jj = i & 255;
  dh0[i] = h0f[bb * H2c + jj];
  dc0[i] = c0f[bb * H2c + jj];
  dh1[i] = h0b[bb * H2c + jj];
  dc1[i] = c0b[bb * H2c + jj];
}

// ---------------- attention: tanh(proj_h + proj_g) . w_out, masked softmax, argmax ----------------
__global__ __launch_bounds__(256)
void attention_kernel(const float* __restrict__ ph, const float* __restrict__ pg,
                      const float* __restrict__ w_out, const float* __restrict__ b_out,
                      const int* __restrict__ x_len, float* __restrict__ outp) {
  int bt   = blockIdx.x;           // b*T + t
  int b    = bt >> 5;              // T = 32
  int lane = threadIdx.x & 31;
  int wave = threadIdx.x >> 5;

  __shared__ float sw[NH];
  __shared__ float sg[NH];
  __shared__ float ssc[SS];
  __shared__ float red[2];

  for (int k = threadIdx.x; k < NH; k += 256) {
    sw[k] = w_out[k];
    sg[k] = pg[(size_t)bt * NH + k];
  }
  __syncthreads();

  float bo = b_out[0];
  for (int s = wave; s < SS; s += 8) {
    const float* phr = ph + ((size_t)b * SS + s) * NH;
    float sum = 0.0f;
    for (int k = lane; k < NH; k += 32)
      sum += tanhf(phr[k] + sg[k]) * sw[k];
#pragma unroll
    for (int off = 16; off; off >>= 1) sum += __shfl_xor(sum, off, 32);
    if (lane == 0) ssc[s] = sum + bo;
  }
  __syncthreads();

  int xl = x_len[b];
  if (threadIdx.x < SS && (int)threadIdx.x >= xl) ssc[threadIdx.x] = -100.0f;
  __syncthreads();

  if (threadIdx.x == 0) {
    float mx = ssc[0]; int arg = 0;
    for (int s = 1; s < SS; ++s) if (ssc[s] > mx) { mx = ssc[s]; arg = s; }
    float den = 0.0f;
    for (int s = 0; s < SS; ++s) den += expf(ssc[s] - mx);
    red[0] = mx; red[1] = den;
    outp[(size_t)BB * TT * SS + bt] = (float)arg;  // choices (first max)
  }
  __syncthreads();
  if (threadIdx.x < SS)
    outp[(size_t)bt * SS + threadIdx.x] = expf(ssc[threadIdx.x] - red[0]) / red[1];
}

// ---------------- host ----------------
extern "C" void kernel_launch(void* const* d_in, const int* in_sizes, int n_in,
                              void* d_out, int out_size, void* d_ws, size_t ws_size,
                              hipStream_t stream) {
  (void)in_sizes; (void)n_in; (void)out_size; (void)ws_size;
  const float* x_emb    = (const float*)d_in[0];
  const float* eWih0    = (const float*)d_in[1];
  const float* eWhh0    = (const float*)d_in[2];
  const float* eb0      = (const float*)d_in[3];
  const float* eWih1    = (const float*)d_in[4];
  const float* eWhh1    = (const float*)d_in[5];
  const float* eb1      = (const float*)d_in[6];
  const float* dWih0    = (const float*)d_in[7];
  const float* dWhh0    = (const float*)d_in[8];
  const float* db0      = (const float*)d_in[9];
  const float* dWih1    = (const float*)d_in[10];
  const float* dWhh1    = (const float*)d_in[11];
  const float* db1      = (const float*)d_in[12];
  const float* Wh       = (const float*)d_in[13];
  const float* bh       = (const float*)d_in[14];
  const float* Wg       = (const float*)d_in[15];
  const float* bg       = (const float*)d_in[16];
  const float* w_out    = (const float*)d_in[17];
  const float* b_out    = (const float*)d_in[18];
  const int*   x_len    = (const int*)d_in[19];
  const int*   gt_tok   = (const int*)d_in[20];
  const int*   gt_len   = (const int*)d_in[21];
  float* outp = (float*)d_out;

  // bump allocator over d_ws
  char* base = (char*)d_ws;
  size_t off = 0;
  auto alloc = [&](size_t bytes) -> void* {
    void* p = base + off;
    off = (off + bytes + 255) & ~(size_t)255;
    return p;
  };
  // f16 weights / activations
  _Float16* wWih0  = (_Float16*)alloc((size_t)2 * 1024 * NWP * 2);
  _Float16* wWhh0  = (_Float16*)alloc((size_t)2 * 1024 * H2c * 2);
  _Float16* wWih1  = (_Float16*)alloc((size_t)2 * 1024 * NH  * 2);
  _Float16* wWhh1  = (_Float16*)alloc((size_t)2 * 1024 * H2c * 2);
  _Float16* wDWhh0 = (_Float16*)alloc((size_t)2048 * NH * 2);
  _Float16* wDWih1 = (_Float16*)alloc((size_t)2048 * NH * 2);
  _Float16* wDWhh1 = (_Float16*)alloc((size_t)2048 * NH * 2);
  _Float16* wWhF   = (_Float16*)alloc((size_t)NH * NH * 2);
  _Float16* wWgF   = (_Float16*)alloc((size_t)NH * NH * 2);
  _Float16* Xp     = (_Float16*)alloc((size_t)BB * SS * NWP * 2);
  _Float16* x1     = (_Float16*)alloc((size_t)BB * SS * NH * 2);
  _Float16* henc   = (_Float16*)alloc((size_t)BB * SS * NH * 2);
  _Float16* d0buf  = (_Float16*)alloc((size_t)BB * TT * NH * 2);
  _Float16* gsbuf  = (_Float16*)alloc((size_t)BB * TT * NH * 2);
  // f32 scratch (gA/gB reused across phases; stream order guarantees safety)
  float* gA    = (float*)alloc((size_t)BB * SS * 1024 * 4);  // == 2048*2048 floats
  float* gB    = (float*)alloc((size_t)BB * SS * 1024 * 4);
  float* projh = (float*)alloc((size_t)BB * SS * NH * 4);
  float* projg = (float*)alloc((size_t)BB * TT * NH * 4);
  float* h0f = (float*)alloc((size_t)BB * H2c * 4);
  float* c0f = (float*)alloc((size_t)BB * H2c * 4);
  float* h0b = (float*)alloc((size_t)BB * H2c * 4);
  float* c0b = (float*)alloc((size_t)BB * H2c * 4);
  float* dh0 = (float*)alloc((size_t)BB * NH * 4);
  float* dc0 = (float*)alloc((size_t)BB * NH * 4);
  float* dh1 = (float*)alloc((size_t)BB * NH * 4);
  float* dc1 = (float*)alloc((size_t)BB * NH * 4);

  auto conv = [&](const float* s, _Float16* d, int rows, int sk, int dk) {
    int nel = rows * dk;
    convert_pad_kernel<<<(nel + 255) / 256, 256, 0, stream>>>(s, d, rows, sk, dk);
  };
  auto gemm = [&](const _Float16* A, const _Float16* W, const float* bias, float* C,
                  int M, int N, int K, int lda, int ldw, int ldc) {
    int tiles = (M >> 4) * (N >> 4);
    wmma_gemm_bias_kernel<<<(tiles + 7) / 8, 256, 0, stream>>>(A, W, bias, C, M, N, K, lda, ldw, ldc);
  };
  auto recur = [&](const float* inG, const _Float16* Whh, const float* h0, const float* c0,
                   const int* lens, _Float16* out, int ldo, int col0,
                   float* hF, float* cF, int S, int H, int rev) {
    lstm_recur_kernel<<<BB / 16, (H / 16) * 32, 0, stream>>>(
        inG, Whh, h0, c0, lens, out, ldo, col0, hF, cF, S, H, rev);
  };

  // 1) convert weights + padded input embedding to f16
  conv(eWih0, wWih0, 2 * 1024, NW, NWP);
  conv(eWhh0, wWhh0, 2 * 1024, H2c, H2c);
  conv(eWih1, wWih1, 2 * 1024, NH, NH);
  conv(eWhh1, wWhh1, 2 * 1024, H2c, H2c);
  conv(dWhh0, wDWhh0, 2048, NH, NH);
  conv(dWih1, wDWih1, 2048, NH, NH);
  conv(dWhh1, wDWhh1, 2048, NH, NH);
  conv(Wh, wWhF, NH, NH, NH);
  conv(Wg, wWgF, NH, NH, NH);
  conv(x_emb, Xp, BB * SS, NW, NWP);

  // 2) encoder layer0 input gates (bias folded)
  gemm(Xp, wWih0,                 eb0,        gA, BB * SS, 1024, NWP, NWP, NWP, 1024);
  gemm(Xp, wWih0 + 1024 * NWP,    eb0 + 1024, gB, BB * SS, 1024, NWP, NWP, NWP, 1024);

  // 3) encoder layer0 recurrences (fwd + bwd), outputs concat into x1
  recur(gA, wWhh0,              nullptr, nullptr, x_len, x1, NH, 0,   h0f, c0f, SS, H2c, 0);
  recur(gB, wWhh0 + 1024 * H2c, nullptr, nullptr, x_len, x1, NH, 256, h0b, c0b, SS, H2c, 1);

  // 4) encoder layer1 input gates
  gemm(x1, wWih1,              eb1,        gA, BB * SS, 1024, NH, NH, NH, 1024);
  gemm(x1, wWih1 + 1024 * NH,  eb1 + 1024, gB, BB * SS, 1024, NH, NH, NH, 1024);

  // 5) encoder layer1 recurrences -> h_enc
  recur(gA, wWhh1,              nullptr, nullptr, x_len, henc, NH, 0,   nullptr, nullptr, SS, H2c, 0);
  recur(gB, wWhh1 + 1024 * H2c, nullptr, nullptr, x_len, henc, NH, 256, nullptr, nullptr, SS, H2c, 1);

  // 6) decoder initial states: concat(h,h)
  build_dec_init_kernel<<<(BB * NH + 255) / 256, 256, 0, stream>>>(
      h0f, c0f, h0b, c0b, dh0, dc0, dh1, dc1);

  // 7) decoder layer0 input gates: one-hot -> column gather of dec_Wih0 (+ bias)
  {
    int nel = BB * TT * 4 * NH;
    dec_gather_kernel<<<(nel + 255) / 256, 256, 0, stream>>>(dWih0, db0, gt_tok, gB);
  }

  // 8) decoder layer0 recurrence -> d0
  recur(gB, wDWhh0, dh0, dc0, gt_len, d0buf, NH, 0, nullptr, nullptr, TT, NH, 0);

  // 9) decoder layer1 input gates
  gemm(d0buf, wDWih1, db1, gA, BB * TT, 2048, NH, NH, NH, 2048);

  // 10) decoder layer1 recurrence -> g_s
  recur(gA, wDWhh1, dh1, dc1, gt_len, gsbuf, NH, 0, nullptr, nullptr, TT, NH, 0);

  // 11) attention projections
  gemm(henc,  wWhF, bh, projh, BB * SS, NH, NH, NH, NH, NH);
  gemm(gsbuf, wWgF, bg, projg, BB * TT, NH, NH, NH, NH, NH);

  // 12) scores + masked softmax + argmax -> d_out
  attention_kernel<<<BB * TT, 256, 0, stream>>>(projh, projg, w_out, b_out, x_len, outp);
}
